// MultiHeadGATLayer_71751723647462
// MI455X (gfx1250) — matrix operand order repaired and checked by reading, not compiled, not run on hardware
//
#include <hip/hip_runtime.h>

#define F_IN 128
#define CH   64
#define NH   4
#define NCOL 320      // 256 cols (W) + 64 cols (res_W), fused GEMM
#define NB   8
#define EPS_GN 1e-5f
#define NEG_SLOPE 0.2f

typedef __attribute__((ext_vector_type(16))) __bf16 v16bf;
typedef __attribute__((ext_vector_type(8)))  float  v8f;

// ---------- helpers ----------
__device__ __forceinline__ unsigned short f2bf(float f) {
  unsigned u = __float_as_uint(f);
  u += 0x7FFFu + ((u >> 16) & 1u);       // round-to-nearest-even
  return (unsigned short)(u >> 16);
}
// monotone float -> u32 key (for atomicMax over signed floats)
__device__ __forceinline__ unsigned fkey(float f) {
  unsigned u = __float_as_uint(f);
  return (u & 0x80000000u) ? ~u : (u | 0x80000000u);
}
__device__ __forceinline__ float funkey(unsigned k) {
  unsigned u = (k & 0x80000000u) ? (k & 0x7FFFFFFFu) : ~k;
  return __uint_as_float(u);
}
__device__ __forceinline__ float leaky(float a) {
  return a > 0.f ? a : NEG_SLOPE * a;
}

// ---------- precision prep: fp32 -> bf16 ----------
__global__ void k_cvt_x(const float* __restrict__ x, unsigned short* __restrict__ xb,
                        long long total) {
  long long i = (long long)blockIdx.x * blockDim.x + threadIdx.x;
  if (i < total) xb[i] = f2bf(x[i]);
}

__global__ void k_prep_w(const float* __restrict__ W, const float* __restrict__ resW,
                         unsigned short* __restrict__ wb) {
  int i = blockIdx.x * blockDim.x + threadIdx.x;      // [0, 320*128)
  if (i >= NCOL * F_IN) return;
  int r = i / F_IN, k = i - r * F_IN;
  float v = (r < NH * CH) ? W[r * F_IN + k] : resW[(r - NH * CH) * F_IN + k];
  wb[i] = f2bf(v);
}

// ---------- fused GEMM: Y[N,320] = X[N,128] * Wcat^T, bf16 WMMA, f32 acc ----------
// One wave per 16x16 output tile; K-loop 128/32 = 4 WMMAs.
__global__ void k_gemm(const unsigned short* __restrict__ Xb,
                       const unsigned short* __restrict__ Wb,
                       float* __restrict__ Y, int nRows) {
  const int lane = threadIdx.x & 31;
  const int wave = (int)((blockIdx.x * (long long)blockDim.x + threadIdx.x) >> 5);
  const int colTiles = NCOL / 16;                     // 20
  const int rowTile = wave / colTiles;
  const int colTile = wave - rowTile * colTiles;
  if (rowTile * 16 >= nRows) return;                  // wave-uniform: EXEC stays all-1s
  const int half = lane >> 4;
  const int l15  = lane & 15;
  const size_t m = (size_t)rowTile * 16 + l15;        // A row this lane owns
  const size_t j = (size_t)colTile * 16 + l15;        // B column this lane owns
  const unsigned short* xrow = Xb + m * F_IN;
  const unsigned short* wrow = Wb + j * F_IN;         // B[k][j] = Wcat[j][k]
  v8f acc = {};
  #pragma unroll
  for (int k0 = 0; k0 < F_IN; k0 += 32) {
    union { v16bf v; unsigned u[8]; } A, Bm;
    // A fragment (16x32 bf16): regs 0-3 -> K = half*8 + {0..7}; regs 4-7 -> K = 16 + half*8 + {0..7}
    #pragma unroll
    for (int g = 0; g < 8; ++g) {
      int ka = (g < 4) ? (half * 8 + 2 * g) : (16 + half * 8 + 2 * (g - 4));
      A.u[g] = *(const unsigned*)(xrow + k0 + ka);
      // B fragment (32x16 bf16): same K order as A: regs 0-3 -> K = half*8 + {0..7}; regs 4-7 -> K = 16 + half*8 + {0..7}
      int kb = ka;
      Bm.u[g] = *(const unsigned*)(wrow + k0 + kb);
    }
    acc = __builtin_amdgcn_wmma_f32_16x16x32_bf16(false, A.v, false, Bm.v,
                                                  (short)0, acc, false, false);
  }
  // D layout: reg r -> row = r + half*8, col = l15
  #pragma unroll
  for (int r = 0; r < 8; ++r) {
    size_t mm = (size_t)rowTile * 16 + r + half * 8;
    Y[mm * NCOL + (size_t)colTile * 16 + l15] = acc[r];
  }
}

// ---------- attention logits: a_src/a_dst [N,H] ----------
__global__ void k_attn_dots(const float* __restrict__ Y,
                            const float* __restrict__ att_src,
                            const float* __restrict__ att_dst,
                            float* __restrict__ asrc, float* __restrict__ adst,
                            long long n) {
  long long i = (long long)blockIdx.x * blockDim.x + threadIdx.x;  // (node, head)
  if (i >= n * NH) return;
  long long node = i / NH;
  int h = (int)(i - node * NH);
  const float* xr = Y + (size_t)node * NCOL + h * CH;
  float s = 0.f, d = 0.f;
  #pragma unroll
  for (int c = 0; c < CH; ++c) {
    s += xr[c] * att_src[h * CH + c];
    d += xr[c] * att_dst[h * CH + c];
  }
  asrc[i] = s;
  adst[i] = d;
}

// ---------- segment softmax pass 1: per-dst max (u32-key atomicMax) ----------
__global__ void k_edge_max(const long long* __restrict__ ei,
                           const float* __restrict__ asrc, const float* __restrict__ adst,
                           unsigned* __restrict__ amax, long long E, long long Etot) {
  long long e = (long long)blockIdx.x * blockDim.x + threadIdx.x;
  if (e >= Etot) return;
  int src, dst;
  if (e < E) { src = (int)ei[e]; dst = (int)ei[E + e]; } else { src = dst = (int)(e - E); }
  #pragma unroll
  for (int h = 0; h < NH; ++h) {
    float a = leaky(asrc[src * NH + h] + adst[dst * NH + h]);
    atomicMax(&amax[dst * NH + h], fkey(a));
  }
}

// ---------- pass 2: per-dst sum of exp ----------
__global__ void k_edge_sum(const long long* __restrict__ ei,
                           const float* __restrict__ asrc, const float* __restrict__ adst,
                           const unsigned* __restrict__ amax, float* __restrict__ denom,
                           long long E, long long Etot) {
  long long e = (long long)blockIdx.x * blockDim.x + threadIdx.x;
  if (e >= Etot) return;
  int src, dst;
  if (e < E) { src = (int)ei[e]; dst = (int)ei[E + e]; } else { src = dst = (int)(e - E); }
  #pragma unroll
  for (int h = 0; h < NH; ++h) {
    float a = leaky(asrc[src * NH + h] + adst[dst * NH + h]);
    atomicAdd(&denom[dst * NH + h], __expf(a - funkey(amax[dst * NH + h])));
  }
}

// ---------- pass 3: aggregate with head-mean folded in (one wave per edge) ----------
// agg[dst][c] += sum_h (coef_h / H) * xl[src][h][c]; xl = Y columns [0,256)
__global__ void k_edge_agg(const long long* __restrict__ ei,
                           const float* __restrict__ asrc, const float* __restrict__ adst,
                           const unsigned* __restrict__ amax, const float* __restrict__ denom,
                           const float* __restrict__ Y, float* __restrict__ agg,
                           long long E, long long Etot) {
  long long t = (long long)blockIdx.x * blockDim.x + threadIdx.x;
  long long e = t >> 5;
  int lane = (int)(t & 31);
  if (e >= Etot) return;
  int src, dst;
  if (e < E) { src = (int)ei[e]; dst = (int)ei[E + e]; } else { src = dst = (int)(e - E); }
  float coef[NH];
  #pragma unroll
  for (int h = 0; h < NH; ++h) {
    float a = leaky(asrc[src * NH + h] + adst[dst * NH + h]);
    float ex = __expf(a - funkey(amax[dst * NH + h]));
    coef[h] = ex / denom[dst * NH + h] * (1.0f / NH);
  }
  const float* xs = Y + (size_t)src * NCOL;
  #pragma unroll
  for (int c0 = 0; c0 < CH; c0 += 32) {
    int c = c0 + lane;
    float v = 0.f;
    #pragma unroll
    for (int h = 0; h < NH; ++h) v += coef[h] * xs[h * CH + c];
    atomicAdd(&agg[(size_t)dst * CH + c], v);
  }
}

// ---------- h = agg + bias_gat + residual; accumulate GraphNorm sums ----------
__global__ void k_combine(const float* __restrict__ Y, const float* __restrict__ bias_gat,
                          const float* __restrict__ res_b, const long long* __restrict__ batch,
                          float* __restrict__ agg, float* __restrict__ sum_bc,
                          float* __restrict__ cnt, long long n) {
  long long i = (long long)blockIdx.x * blockDim.x + threadIdx.x;
  if (i >= n * CH) return;
  long long node = i / CH;
  int c = (int)(i - node * CH);
  float h = agg[i] + bias_gat[c] + Y[(size_t)node * NCOL + NH * CH + c] + res_b[c];
  agg[i] = h;                                   // h overwrites agg in-place
  int b = (int)batch[node];
  atomicAdd(&sum_bc[b * CH + c], h);
  if (c == 0) atomicAdd(&cnt[b], 1.0f);
}

__global__ void k_mean(const float* __restrict__ sum_bc, const float* __restrict__ cnt,
                       float* __restrict__ mean_bc) {
  int i = blockIdx.x * blockDim.x + threadIdx.x;
  if (i < NB * CH) mean_bc[i] = sum_bc[i] / cnt[i / CH];
}

__global__ void k_var(const float* __restrict__ hbuf, const long long* __restrict__ batch,
                      const float* __restrict__ mean_bc, const float* __restrict__ gms,
                      float* __restrict__ var_bc, long long n) {
  long long i = (long long)blockIdx.x * blockDim.x + threadIdx.x;
  if (i >= n * CH) return;
  long long node = i / CH;
  int c = (int)(i - node * CH);
  int b = (int)batch[node];
  float cen = hbuf[i] - mean_bc[b * CH + c] * gms[c];
  atomicAdd(&var_bc[b * CH + c], cen * cen);
}

__global__ void k_varfin(float* __restrict__ var_bc, const float* __restrict__ cnt) {
  int i = blockIdx.x * blockDim.x + threadIdx.x;
  if (i < NB * CH) var_bc[i] = var_bc[i] / cnt[i / CH];
}

__global__ void k_final(const float* __restrict__ hbuf, const long long* __restrict__ batch,
                        const float* __restrict__ mean_bc, const float* __restrict__ var_bc,
                        const float* __restrict__ gms, const float* __restrict__ gw,
                        const float* __restrict__ gb, float* __restrict__ out, long long n) {
  long long i = (long long)blockIdx.x * blockDim.x + threadIdx.x;
  if (i >= n * CH) return;
  long long node = i / CH;
  int c = (int)(i - node * CH);
  int b = (int)batch[node];
  float cen = hbuf[i] - mean_bc[b * CH + c] * gms[c];
  float nrm = gw[c] * cen * rsqrtf(var_bc[b * CH + c] + EPS_GN) + gb[c];
  // tanh-approx GELU (jax.nn.gelu default)
  float t = 0.7978845608028654f * (nrm + 0.044715f * nrm * nrm * nrm);
  out[i] = 0.5f * nrm * (1.0f + tanhf(t));
}

extern "C" void kernel_launch(void* const* d_in, const int* in_sizes, int n_in,
                              void* d_out, int out_size, void* d_ws, size_t ws_size,
                              hipStream_t stream) {
  const float*     x        = (const float*)d_in[0];
  const long long* ei       = (const long long*)d_in[1];   // int64 [2,E]
  const long long* batch    = (const long long*)d_in[2];   // int64 [N]
  const float*     W        = (const float*)d_in[3];
  const float*     att_src  = (const float*)d_in[4];
  const float*     att_dst  = (const float*)d_in[5];
  const float*     bias_gat = (const float*)d_in[6];
  const float*     res_W    = (const float*)d_in[7];
  const float*     res_b    = (const float*)d_in[8];
  const float*     gn_w     = (const float*)d_in[9];
  const float*     gn_b     = (const float*)d_in[10];
  const float*     gn_ms    = (const float*)d_in[11];

  const long long n    = (long long)in_sizes[0] / F_IN;
  const long long E    = (long long)in_sizes[1] / 2;
  const long long Etot = E + n;

  // ---- workspace carve-up (256B aligned) ----
  char* ws = (char*)d_ws;
  size_t off = 0;
  auto take = [&](size_t bytes) -> char* {
    char* p = ws + off;
    off = (off + bytes + 255) & ~(size_t)255;
    return p;
  };
  unsigned short* xb      = (unsigned short*)take((size_t)n * F_IN * 2);
  unsigned short* wb      = (unsigned short*)take((size_t)NCOL * F_IN * 2);
  float*          Y       = (float*)take((size_t)n * NCOL * 4);
  float*          asrc    = (float*)take((size_t)n * NH * 4);
  float*          adst    = (float*)take((size_t)n * NH * 4);
  unsigned*       amax    = (unsigned*)take((size_t)n * NH * 4);
  float*          denom   = (float*)take((size_t)n * NH * 4);
  float*          agg     = (float*)take((size_t)n * CH * 4);
  float*          sum_bc  = (float*)take((size_t)NB * CH * 4);
  float*          mean_bc = (float*)take((size_t)NB * CH * 4);
  float*          var_bc  = (float*)take((size_t)NB * CH * 4);
  float*          cnt     = (float*)take((size_t)NB * 4);
  (void)ws_size; (void)n_in; (void)out_size;

  // capture-safe zero-init of accumulators (amax key 0 < key(-inf))
  hipMemsetAsync(amax,   0, (size_t)n * NH * 4, stream);
  hipMemsetAsync(denom,  0, (size_t)n * NH * 4, stream);
  hipMemsetAsync(agg,    0, (size_t)n * CH * 4, stream);
  hipMemsetAsync(sum_bc, 0, (size_t)NB * CH * 4, stream);
  hipMemsetAsync(var_bc, 0, (size_t)NB * CH * 4, stream);
  hipMemsetAsync(cnt,    0, (size_t)NB * 4, stream);

  const int TB = 256;
  k_cvt_x<<<(unsigned)((n * F_IN + TB - 1) / TB), TB, 0, stream>>>(x, xb, n * F_IN);
  k_prep_w<<<(NCOL * F_IN + TB - 1) / TB, TB, 0, stream>>>(W, res_W, wb);

  long long waves = ((n + 15) / 16) * (NCOL / 16);
  k_gemm<<<(unsigned)((waves * 32 + TB - 1) / TB), TB, 0, stream>>>(xb, wb, Y, (int)n);

  k_attn_dots<<<(unsigned)((n * NH + TB - 1) / TB), TB, 0, stream>>>(Y, att_src, att_dst,
                                                                     asrc, adst, n);
  k_edge_max<<<(unsigned)((Etot + TB - 1) / TB), TB, 0, stream>>>(ei, asrc, adst, amax, E, Etot);
  k_edge_sum<<<(unsigned)((Etot + TB - 1) / TB), TB, 0, stream>>>(ei, asrc, adst, amax, denom,
                                                                  E, Etot);
  k_edge_agg<<<(unsigned)((Etot * 32 + TB - 1) / TB), TB, 0, stream>>>(ei, asrc, adst, amax,
                                                                       denom, Y, agg, E, Etot);
  k_combine<<<(unsigned)((n * CH + TB - 1) / TB), TB, 0, stream>>>(Y, bias_gat, res_b, batch,
                                                                   agg, sum_bc, cnt, n);
  k_mean<<<(NB * CH + TB - 1) / TB, TB, 0, stream>>>(sum_bc, cnt, mean_bc);
  k_var<<<(unsigned)((n * CH + TB - 1) / TB), TB, 0, stream>>>(agg, batch, mean_bc, gn_ms,
                                                               var_bc, n);
  k_varfin<<<(NB * CH + TB - 1) / TB, TB, 0, stream>>>(var_bc, cnt);
  k_final<<<(unsigned)((n * CH + TB - 1) / TB), TB, 0, stream>>>(agg, batch, mean_bc, var_bc,
                                                                 gn_ms, gn_w, gn_b,
                                                                 (float*)d_out, n);
}
